// LSTMLanguageModel_1297080123447
// MI455X (gfx1250) — compile-verified, matrix operand order checked by reading, not run on hardware
//
#include <hip/hip_runtime.h>
#include <math.h>

// LSTM LM: embed -> 2-layer LSTM (bf16 WMMA GEMMs, f32 cell state) -> vocab
// projection (bf16 WMMA, 2x2 register blocking) -> online log-softmax.
// Outputs: logp[B,T,V], hT[L,B,H], cT[L,B,H].
// GEMM inner loops use __builtin_amdgcn_sched_group_barrier to pin a
// load/WMMA software pipeline (chunk-1 loads issue between chunk-0 WMMAs),
// so waitcnt thresholds become nonzero and memory overlaps the matrix pipe.

#define B_  32
#define T_  128
#define V_  50257
#define H_  1024
#define L_  2
#define G4H (4 * H_)   // 4096
#define BT  (B_ * T_)  // 4096

typedef __attribute__((ext_vector_type(8)))  __bf16 v8bf;
typedef __attribute__((ext_vector_type(16))) __bf16 v16bf;
typedef __attribute__((ext_vector_type(8)))  float  v8f;

// Scheduling-intent groups (AMDGPU): 0x020 = VMEM read, 0x008 = MFMA/WMMA.
#if defined(__has_builtin)
#if __has_builtin(__builtin_amdgcn_sched_group_barrier)
#define SCHED_VMEM(n) __builtin_amdgcn_sched_group_barrier(0x020, (n), 0)
#define SCHED_WMMA(n) __builtin_amdgcn_sched_group_barrier(0x008, (n), 0)
#endif
#endif
#ifndef SCHED_VMEM
#define SCHED_VMEM(n)
#define SCHED_WMMA(n)
#endif

// ---------------------------------------------------------------------------
// Load one 16x32 bf16 WMMA operand tile (A layout; B uses the same per-lane
// layout with "row" = output column of W). Per ISA 7.12.2: lanes 0-15 hold
// K=0..7 (VGPR0-3) and K=16..23 (VGPR4-7); lanes 16-31 hold K=8..15 / K=24..31.
// -> per lane: two contiguous 8-element (16B) chunks at k0+half*8 and k0+16+half*8.
// ---------------------------------------------------------------------------
__device__ __forceinline__ v16bf load_tile_bf16(const __bf16* __restrict__ base,
                                                long rowStride, int row, int k0) {
  const int lane = threadIdx.x & 31;
  const int half = lane >> 4;
  const __bf16* p = base + (long)row * rowStride + k0 + half * 8;
  const v8bf lo = *(const v8bf*)(p);
  const v8bf hi = *(const v8bf*)(p + 16);
  v16bf out;
#pragma unroll
  for (int i = 0; i < 8; ++i) { out[i] = lo[i]; out[i + 8] = hi[i]; }
  return out;
}

__device__ __forceinline__ float sigmoidf_(float x) {
  return 1.0f / (1.0f + __expf(-x));
}

#define WMMA_BF16(A, Bm, C) \
  __builtin_amdgcn_wmma_f32_16x16x32_bf16(false, (A), false, (Bm), (short)0, (C), false, false)

// ---------------------------------------------------------------------------
// fp32 -> bf16 bulk convert (weights). Grid-stride.
// ---------------------------------------------------------------------------
__global__ void k_f32_to_bf16(const float* __restrict__ src,
                              __bf16* __restrict__ dst, long n) {
  long i = (long)blockIdx.x * blockDim.x + threadIdx.x;
  const long stride = (long)gridDim.x * blockDim.x;
  for (; i < n; i += stride) dst[i] = (__bf16)src[i];
}

// Embedding gather: one block per (b,t) row. x_bf is [B,T,H] bf16.
__global__ void k_embed(const int* __restrict__ tokens,
                        const float* __restrict__ emb,
                        __bf16* __restrict__ x_bf) {
  const int bt  = blockIdx.x;
  const int tok = tokens[bt];
  const float* e = emb + (long)tok * H_;
  __bf16* o = x_bf + (long)bt * H_;
  for (int i = threadIdx.x; i < H_; i += blockDim.x) o[i] = (__bf16)e[i];
}

// Init h/c state from h0/c0; hbf slot 0 of each layer gets bf16(h0).
__global__ void k_init_state(const float* __restrict__ h0,
                             const float* __restrict__ c0,
                             float* __restrict__ h_state,
                             float* __restrict__ c_state,
                             __bf16* __restrict__ hbf) {   // [L][2][B*H]
  const long n = (long)L_ * B_ * H_;
  long i = (long)blockIdx.x * blockDim.x + threadIdx.x;
  if (i < n) {
    h_state[i] = h0[i];
    c_state[i] = c0[i];
    const long l = i / (B_ * H_);
    const long r = i % (B_ * H_);
    hbf[(l * 2) * (long)(B_ * H_) + r] = (__bf16)h0[i];   // slot 0 (t=0 reads cur=0)
  }
}

// ---------------------------------------------------------------------------
// One LSTM timestep for one layer.
//   grid.x = H/16 = 64 (one block per 16 hidden columns)
//   block  = 128 threads = 4 waves; wave g computes gate g (i,f,g,o)
// Each wave: gates[0:32, j0:j0+16] = xA@Wih^T + hA@Whh^T, K=1024 via
// v_wmma_f32_16x16x32_bf16 (two M-tiles). Gates -> LDS -> fused cell update.
// Weights (33.5 MB bf16 total) stay resident in the 192 MB L2 across steps.
// Pipeline per iteration: 12 loads | 2 WMMA | 6 loads | 2 WMMA | 6 loads | 4 WMMA.
// ---------------------------------------------------------------------------
__global__ void __launch_bounds__(128)
k_lstm_step(const __bf16* __restrict__ xA, long xStride,   // [B,*] layer input
            const __bf16* __restrict__ hA,                 // [B,H] prev hidden (bf16)
            const __bf16* __restrict__ Wih,                // [4H,H]
            const __bf16* __restrict__ Whh,                // [4H,H]
            const float* __restrict__ bih,
            const float* __restrict__ bhh,
            float* __restrict__ c_state,                   // [B,H]
            float* __restrict__ h_state,                   // [B,H] fp32 (for hT)
            __bf16* __restrict__ h_bf_out,                 // [B,H] next-step input
            __bf16* __restrict__ extra_out, long extraStride) { // outs (layer 1)
  __shared__ float G[4][32][16];

  const int gate = threadIdx.x >> 5;        // wave id = gate id
  const int lane = threadIdx.x & 31;
  const int half = lane >> 4;
  const int r    = lane & 15;
  const int j0   = blockIdx.x * 16;
  const int wrow = gate * H_ + j0 + r;      // W row == output column for B operand

  v8f acc0 = {};  // rows 0..15
  v8f acc1 = {};  // rows 16..31

  for (int k0 = 0; k0 < H_; k0 += 64) {
    // ---- chunk 0 loads (12 b128) ----
    const v16bf bi0  = load_tile_bf16(Wih, H_, wrow, k0);
    const v16bf bh0  = load_tile_bf16(Whh, H_, wrow, k0);
    const v16bf ax00 = load_tile_bf16(xA, xStride, r,      k0);
    const v16bf ax10 = load_tile_bf16(xA, xStride, 16 + r, k0);
    const v16bf ah00 = load_tile_bf16(hA, H_,      r,      k0);
    const v16bf ah10 = load_tile_bf16(hA, H_,      16 + r, k0);
    // ---- chunk 1 loads (12 b128) ----
    const v16bf bi1  = load_tile_bf16(Wih, H_, wrow, k0 + 32);
    const v16bf bh1  = load_tile_bf16(Whh, H_, wrow, k0 + 32);
    const v16bf ax01 = load_tile_bf16(xA, xStride, r,      k0 + 32);
    const v16bf ax11 = load_tile_bf16(xA, xStride, 16 + r, k0 + 32);
    const v16bf ah01 = load_tile_bf16(hA, H_,      r,      k0 + 32);
    const v16bf ah11 = load_tile_bf16(hA, H_,      16 + r, k0 + 32);
    // ---- WMMAs ----
    acc0 = WMMA_BF16(ax00, bi0, acc0);
    acc1 = WMMA_BF16(ax10, bi0, acc1);
    acc0 = WMMA_BF16(ah00, bh0, acc0);
    acc1 = WMMA_BF16(ah10, bh0, acc1);
    acc0 = WMMA_BF16(ax01, bi1, acc0);
    acc1 = WMMA_BF16(ax11, bi1, acc1);
    acc0 = WMMA_BF16(ah01, bh1, acc0);
    acc1 = WMMA_BF16(ah11, bh1, acc1);
    // Pin the pipeline: chunk-1 loads issue between chunk-0 WMMAs, so the
    // wait before each WMMA group leaves later loads in flight.
    SCHED_VMEM(12);   // chunk-0 loads
    SCHED_WMMA(2);
    SCHED_VMEM(6);    // half of chunk-1 loads
    SCHED_WMMA(2);
    SCHED_VMEM(6);    // rest of chunk-1 loads
    SCHED_WMMA(4);
  }

  const float bias = bih[wrow] + bhh[wrow];  // depends only on column
#pragma unroll
  for (int v = 0; v < 8; ++v) {
    // C/D layout: lanes 0-15 -> M=v, lanes 16-31 -> M=8+v; N = lane&15
    G[gate][half * 8 + v][r]      = acc0[v] + bias;
    G[gate][16 + half * 8 + v][r] = acc1[v] + bias;
  }
  __syncthreads();

  for (int idx = threadIdx.x; idx < 32 * 16; idx += 128) {
    const int m = idx >> 4;         // batch row
    const int n = idx & 15;         // column within tile
    const int j = j0 + n;
    const float ig = sigmoidf_(G[0][m][n]);
    const float fg = sigmoidf_(G[1][m][n]);
    const float gg = tanhf(G[2][m][n]);
    const float og = sigmoidf_(G[3][m][n]);
    const long off = (long)m * H_ + j;
    const float cn = fg * c_state[off] + ig * gg;
    const float hn = og * tanhf(cn);
    c_state[off]  = cn;
    h_state[off]  = hn;
    h_bf_out[off] = (__bf16)hn;
    if (extra_out) extra_out[(long)m * extraStride + j] = (__bf16)hn;
  }
}

// ---------------------------------------------------------------------------
// Vocab projection: logits[BT, V] = outs[BT,H] @ W_out[V,H]^T + b_out.
// 2x2 register blocking per wave (32x32 output block), k-loop unrolled x2.
// Pipeline per iteration: 8 loads | 2 WMMA | 4 loads | 2 WMMA | 4 loads | 4 WMMA.
//   grid = (ceil(V/32)=1571, BT/256=16); block = 256 = 8 waves.
//   wave w: rows m0 = by*256 + w*32; cols n0 = bx*32.
// W_out streamed ~16x (=1.6 GB); W rows prefetched 512 k ahead (global_prefetch_b8).
// ---------------------------------------------------------------------------
__global__ void __launch_bounds__(256)
k_proj(const __bf16* __restrict__ A,      // [BT, H]
       const __bf16* __restrict__ W,      // [V, H]
       const float* __restrict__ bout,    // [V]
       float* __restrict__ logits) {      // [BT, V]
  const int wave = threadIdx.x >> 5;
  const int lane = threadIdx.x & 31;
  const int half = lane >> 4;
  const int r    = lane & 15;
  const int n0   = blockIdx.x * 32;
  const int m0   = blockIdx.y * 256 + wave * 32;
  const int vr0  = n0 + r;               // B rows (output columns), tile 0/1
  const int vr1  = n0 + 16 + r;
  const int vc0  = (vr0 < V_) ? vr0 : (V_ - 1);   // clamp loads at V edge
  const int vc1  = (vr1 < V_) ? vr1 : (V_ - 1);

  v8f acc00 = {}, acc01 = {}, acc10 = {}, acc11 = {};

  for (int k0 = 0; k0 < H_; k0 += 64) {
    // ---- chunk 0 loads (8 b128) ----
    const v16bf a0_0 = load_tile_bf16(A, H_, m0 + r,      k0);
    const v16bf a1_0 = load_tile_bf16(A, H_, m0 + 16 + r, k0);
    const v16bf b0_0 = load_tile_bf16(W, H_, vc0, k0);
    const v16bf b1_0 = load_tile_bf16(W, H_, vc1, k0);
    // ---- chunk 1 loads (8 b128) ----
    const v16bf a0_1 = load_tile_bf16(A, H_, m0 + r,      k0 + 32);
    const v16bf a1_1 = load_tile_bf16(A, H_, m0 + 16 + r, k0 + 32);
    const v16bf b0_1 = load_tile_bf16(W, H_, vc0, k0 + 32);
    const v16bf b1_1 = load_tile_bf16(W, H_, vc1, k0 + 32);
    // ---- WMMAs ----
    acc00 = WMMA_BF16(a0_0, b0_0, acc00);
    acc01 = WMMA_BF16(a0_0, b1_0, acc01);
    acc10 = WMMA_BF16(a1_0, b0_0, acc10);
    acc11 = WMMA_BF16(a1_0, b1_0, acc11);
    acc00 = WMMA_BF16(a0_1, b0_1, acc00);
    acc01 = WMMA_BF16(a0_1, b1_1, acc01);
    acc10 = WMMA_BF16(a1_1, b0_1, acc10);
    acc11 = WMMA_BF16(a1_1, b1_1, acc11);
    // Prefetch the streaming W_out rows 512 k (8 unrolled iterations) ahead
    // (gfx1250 global_prefetch_b8; speculative, OOB-safe).
    __builtin_prefetch(W + (long)vc0 * H_ + k0 + 512, 0, 0);
    __builtin_prefetch(W + (long)vc1 * H_ + k0 + 512, 0, 0);
    // Pin the pipeline.
    SCHED_VMEM(8);    // chunk-0 loads
    SCHED_WMMA(2);
    SCHED_VMEM(4);    // half of chunk-1 loads
    SCHED_WMMA(2);
    SCHED_VMEM(4);    // rest of chunk-1 loads
    SCHED_WMMA(4);
  }

  // Epilogue: write 4 tiles, bounds-checked at the V edge (after all WMMAs).
  const float bb0 = bout[vc0];
  const float bb1 = bout[vc1];
  if (vr0 < V_) {
#pragma unroll
    for (int v = 0; v < 8; ++v) {
      const int m = m0 + half * 8 + v;
      logits[(long)m * V_ + vr0]        = acc00[v] + bb0;
      logits[(long)(m + 16) * V_ + vr0] = acc10[v] + bb0;
    }
  }
  if (vr1 < V_) {
#pragma unroll
    for (int v = 0; v < 8; ++v) {
      const int m = m0 + half * 8 + v;
      logits[(long)m * V_ + vr1]        = acc01[v] + bb1;
      logits[(long)(m + 16) * V_ + vr1] = acc11[v] + bb1;
    }
  }
}

// ---------------------------------------------------------------------------
// In-place row log-softmax over V, ONLINE max+sum first pass (2R+1W instead
// of 3R+1W: saves one full 823 MB read of the logits).
// One block (256 threads) per (b,t) row.
// ---------------------------------------------------------------------------
__global__ void __launch_bounds__(256)
k_logsoftmax(float* __restrict__ data) {
  __shared__ float sm[256];
  __shared__ float ss[256];
  const int tid = threadIdx.x;
  float* p = data + (long)blockIdx.x * V_;

  // Online (max, sum-of-exp) over this thread's strided slice.
  float mx = -INFINITY;
  float sum = 0.0f;
  for (int i = tid; i < V_; i += 256) {
    const float x  = p[i];
    const float nm = fmaxf(mx, x);
    sum = sum * __expf(mx - nm) + __expf(x - nm);
    mx  = nm;
  }
  sm[tid] = mx; ss[tid] = sum;
  __syncthreads();
  for (int s = 128; s > 0; s >>= 1) {
    if (tid < s) {
      const float m2 = sm[tid + s], s2 = ss[tid + s];
      const float nm = fmaxf(sm[tid], m2);
      ss[tid] = ss[tid] * __expf(sm[tid] - nm) + s2 * __expf(m2 - nm);
      sm[tid] = nm;
    }
    __syncthreads();
  }
  const float lse = sm[0] + __logf(ss[0]);

  for (int i = tid; i < V_; i += 256) p[i] -= lse;
}

// Final hT/cT tail: out_tail[0:LBH]=h_state, out_tail[LBH:2LBH]=c_state.
__global__ void k_copy_state(const float* __restrict__ h_state,
                             const float* __restrict__ c_state,
                             float* __restrict__ out_tail) {
  const int n = L_ * B_ * H_;
  const int i = blockIdx.x * blockDim.x + threadIdx.x;
  if (i < n) {
    out_tail[i]     = h_state[i];
    out_tail[n + i] = c_state[i];
  }
}

// ---------------------------------------------------------------------------
extern "C" void kernel_launch(void* const* d_in, const int* in_sizes, int n_in,
                              void* d_out, int out_size, void* d_ws, size_t ws_size,
                              hipStream_t stream) {
  (void)in_sizes; (void)n_in; (void)out_size; (void)ws_size;

  const int*   tokens = (const int*)  d_in[0];
  const float* emb    = (const float*)d_in[1];
  const float* W_ih   = (const float*)d_in[2];
  const float* W_hh   = (const float*)d_in[3];
  const float* b_ih   = (const float*)d_in[4];
  const float* b_hh   = (const float*)d_in[5];
  const float* W_out  = (const float*)d_in[6];
  const float* b_out  = (const float*)d_in[7];
  const float* h0     = (const float*)d_in[8];
  const float* c0     = (const float*)d_in[9];
  float* out = (float*)d_out;

  // Workspace carve-out (256B aligned). Total ~154 MB.
  char* wsp = (char*)d_ws;
  auto alloc = [&](size_t bytes) -> char* {
    char* p = wsp;
    wsp += (bytes + 255) & ~(size_t)255;
    return p;
  };
  const size_t BH = (size_t)B_ * H_;
  __bf16* Wih_bf  = (__bf16*)alloc((size_t)L_ * G4H * H_ * sizeof(__bf16));
  __bf16* Whh_bf  = (__bf16*)alloc((size_t)L_ * G4H * H_ * sizeof(__bf16));
  __bf16* Wout_bf = (__bf16*)alloc((size_t)V_ * H_ * sizeof(__bf16));
  __bf16* x_bf    = (__bf16*)alloc((size_t)BT * H_ * sizeof(__bf16));
  __bf16* outs_bf = (__bf16*)alloc((size_t)BT * H_ * sizeof(__bf16));
  __bf16* hbf     = (__bf16*)alloc((size_t)L_ * 2 * BH * sizeof(__bf16)); // ping-pong per layer
  float*  h_state = (float*) alloc((size_t)L_ * BH * sizeof(float));
  float*  c_state = (float*) alloc((size_t)L_ * BH * sizeof(float));

  // Weight conversion + embedding + state init.
  k_f32_to_bf16<<<2048, 256, 0, stream>>>(W_ih,  Wih_bf,  (long)L_ * G4H * H_);
  k_f32_to_bf16<<<2048, 256, 0, stream>>>(W_hh,  Whh_bf,  (long)L_ * G4H * H_);
  k_f32_to_bf16<<<8192, 256, 0, stream>>>(W_out, Wout_bf, (long)V_ * H_);
  k_embed<<<BT, 256, 0, stream>>>(tokens, emb, x_bf);
  k_init_state<<<(L_ * B_ * H_ + 255) / 256, 256, 0, stream>>>(h0, c0, h_state, c_state, hbf);

  // Recurrence: 128 steps x 2 layers. h double-buffered (read cur, write nxt).
  for (int t = 0; t < T_; ++t) {
    const int cur = t & 1;
    const int nxt = cur ^ 1;
    // Layer 0: input = embedded tokens at time t (rows = batch, stride T*H).
    k_lstm_step<<<H_ / 16, 128, 0, stream>>>(
        x_bf + (size_t)t * H_, (long)T_ * H_,
        hbf + (0 * 2 + cur) * BH,
        Wih_bf, Whh_bf, b_ih, b_hh,
        c_state, h_state,
        hbf + (0 * 2 + nxt) * BH,
        (__bf16*)nullptr, 0L);
    // Layer 1: input = layer-0 hidden just produced; also writes outs[b,t,:].
    k_lstm_step<<<H_ / 16, 128, 0, stream>>>(
        hbf + (0 * 2 + nxt) * BH, (long)H_,
        hbf + (1 * 2 + cur) * BH,
        Wih_bf + (size_t)G4H * H_, Whh_bf + (size_t)G4H * H_,
        b_ih + G4H, b_hh + G4H,
        c_state + BH, h_state + BH,
        hbf + (1 * 2 + nxt) * BH,
        outs_bf + (size_t)t * H_, (long)T_ * H_);
  }

  // Projection (dominant GEMM) straight into d_out, then in-place log-softmax.
  dim3 pgrid((V_ + 31) / 32, BT / 256);
  k_proj<<<pgrid, 256, 0, stream>>>(outs_bf, Wout_bf, b_out, out);
  k_logsoftmax<<<BT, 256, 0, stream>>>(out);
  k_copy_state<<<(L_ * B_ * H_ + 255) / 256, 256, 0, stream>>>(
      h_state, c_state, out + (size_t)BT * V_);
}
